// RieszFeatureEnhancer_47278999995042
// MI455X (gfx1250) — matrix operand order repaired and checked by reading, not compiled, not run on hardware
//
#include <hip/hip_runtime.h>
#include <hip/hip_bf16.h>

// ---------------------------------------------------------------------------
// RieszFeatureEnhancer for MI455X (gfx1250, wave32, WMMA).
// 2-D FFT / inverse FFT implemented as DFT-matrix GEMMs on the f16 WMMA path
// (v_wmma_f32_16x16x32_f16, f32 accumulate). Primitive: Y = Fv @ Z^T applied
// twice == Fv Z Fv, which makes every B-operand fragment a contiguous 32B
// load from row-major Z (no LDS staging needed).
// ---------------------------------------------------------------------------

typedef __attribute__((ext_vector_type(16))) _Float16 half16;
typedef __attribute__((ext_vector_type(8)))  _Float16 half8;
typedef __attribute__((ext_vector_type(8)))  float    float8;

#define N_DIM 256
#define SLICE 65536   // 256*256 elements per (b,c) slice
#define NSLICE 256    // B*C

__device__ __forceinline__ float8 wmma_f16(half16 a, half16 b, float8 c) {
  // 8 args: (neg_a, A, neg_b, B, c_mod, C, reuse_a, reuse_b)
  return __builtin_amdgcn_wmma_f32_16x16x32_f16(false, a, false, b, (short)0, c,
                                                false, false);
}

__device__ __forceinline__ half16 neg16h(half16 a) {
  half16 r;
#pragma unroll
  for (int i = 0; i < 16; i++) r[i] = -a[i];
  return r;
}

// A fragment (16x32 f16, M = lane&15): lanes<16 hold K = kb+[0..15];
// lanes>=16 hold K = kb+[16..31]  (contiguous, matches B fragment).
__device__ __forceinline__ half16 load_afrag(const _Float16* __restrict__ F,
                                             int mrow, int kb, int lane) {
  const int k0 = kb + ((lane >> 4) & 1) * 16;
  const _Float16* p = F + mrow * N_DIM + k0;
  half8 lo = *(const half8*)(p);
  half8 hi = *(const half8*)(p + 8);
  half16 a;
#pragma unroll
  for (int i = 0; i < 8; i++) { a[i] = lo[i]; a[i + 8] = hi[i]; }
  return a;
}

// ---------------------------------------------------------------------------
// MODE 0: B real (f32 x), forward F.         out: complex f16  (T = F x^T)
// MODE 1: B complex f16, forward F.          out: complex f16  (S = F T^T)
// MODE 2: B complex f16 * (i*spec) on load, conj F. out: complex f16
// MODE 3: B complex f16, conj F.             out: f32 |.|/65536 + 1e-6
// Grid (2,2,NSLICE), block 256 (8 waves); 128x128 tile/block, 32x64/wave.
// ---------------------------------------------------------------------------
template <int MODE>
__global__ void __launch_bounds__(256) dft_gemm(
    const _Float16* __restrict__ Fr, const _Float16* __restrict__ Fi,
    const float* __restrict__ Breal,
    const _Float16* __restrict__ Bzr, const _Float16* __restrict__ Bzi,
    const float* __restrict__ spec,
    _Float16* __restrict__ Or, _Float16* __restrict__ Oi,
    float* __restrict__ Oabs) {
  const int lane = threadIdx.x & 31;
  const int wave = threadIdx.x >> 5;
  const int Moff = blockIdx.x * 128 + (wave & 3) * 32;   // 4 waves along M
  const int Noff = blockIdx.y * 128 + (wave >> 2) * 64;  // 2 waves along N
  const size_t so = (size_t)blockIdx.z * SLICE;

  const float8 zacc = {0.f, 0.f, 0.f, 0.f, 0.f, 0.f, 0.f, 0.f};
  float8 cr[2][4], ci[2][4];
#pragma unroll
  for (int t = 0; t < 2; t++)
#pragma unroll
    for (int u = 0; u < 4; u++) { cr[t][u] = zacc; ci[t][u] = zacc; }

  for (int kb = 0; kb < N_DIM; kb += 32) {
    // A fragments: Ar = Fr, Ai = +/-Fi (conj for inverse stages), and -Ai.
    half16 ar[2], aip[2], ain[2];
#pragma unroll
    for (int t = 0; t < 2; t++) {
      const int mrow = Moff + t * 16 + (lane & 15);
      ar[t] = load_afrag(Fr, mrow, kb, lane);
      half16 fi = load_afrag(Fi, mrow, kb, lane);
      half16 nfi = neg16h(fi);
      if (MODE >= 2) { aip[t] = nfi; ain[t] = fi; }   // A = conj(F) = Fr - i*Fi
      else           { aip[t] = fi;  ain[t] = nfi; }  // A = F
    }
#pragma unroll
    for (int u = 0; u < 4; u++) {
      const int nrow = Noff + u * 16 + (lane & 15);
      const int k0 = kb + ((lane >> 4) & 1) * 16;  // B frag: 16 contiguous K
      half16 br, bi;
      if (MODE == 0) {
        const float* p = Breal + so + (size_t)nrow * N_DIM + k0;
#pragma unroll
        for (int i = 0; i < 16; i++) br[i] = (_Float16)p[i];
      } else if (MODE == 2) {
        // U = (i*m) * S  ->  Ur = -m*Si, Ui = m*Sr   (Riesz maps are imaginary)
        const _Float16* psr = Bzr + so + (size_t)nrow * N_DIM + k0;
        const _Float16* psi = Bzi + so + (size_t)nrow * N_DIM + k0;
        const float* pm = spec + (size_t)nrow * N_DIM + k0;
        half16 sr = *(const half16*)psr;
        half16 si = *(const half16*)psi;
#pragma unroll
        for (int i = 0; i < 16; i++) {
          float m = pm[i];
          br[i] = (_Float16)(-m * (float)si[i]);
          bi[i] = (_Float16)( m * (float)sr[i]);
        }
      } else {
        br = *(const half16*)(Bzr + so + (size_t)nrow * N_DIM + k0);
        bi = *(const half16*)(Bzi + so + (size_t)nrow * N_DIM + k0);
      }
#pragma unroll
      for (int t = 0; t < 2; t++) {
        // Cr += Ar*Br - Ai*Bi ; Ci += Ar*Bi + Ai*Br
        cr[t][u] = wmma_f16(ar[t], br, cr[t][u]);
        ci[t][u] = wmma_f16(aip[t], br, ci[t][u]);
        if (MODE != 0) {
          cr[t][u] = wmma_f16(ain[t], bi, cr[t][u]);
          ci[t][u] = wmma_f16(ar[t], bi, ci[t][u]);
        }
      }
    }
  }

  // C/D layout: VGPR r of lane L -> (M = 8*(L>=16)+r, N = L&15)
  const int mb = ((lane >> 4) & 1) * 8;
  const int nb = lane & 15;
#pragma unroll
  for (int t = 0; t < 2; t++)
#pragma unroll
    for (int u = 0; u < 4; u++)
#pragma unroll
      for (int r = 0; r < 8; r++) {
        const int row = Moff + t * 16 + mb + r;
        const int col = Noff + u * 16 + nb;
        const size_t off = so + (size_t)row * N_DIM + col;
        if (MODE == 3) {
          float vr = cr[t][u][r], vi = ci[t][u][r];
          Oabs[off] = sqrtf(vr * vr + vi * vi) * (1.0f / 65536.0f) + 1e-6f;
        } else {
          Or[off] = (_Float16)cr[t][u][r];
          Oi[off] = (_Float16)ci[t][u][r];
        }
      }
}

// ---------------------------------------------------------------------------
// Init: F matrix (f16), Riesz spectral maps (imag parts, with transposed
// trker to match _frft2d's internal transposition), zero gf accumulators.
// ---------------------------------------------------------------------------
__global__ void __launch_bounds__(256) init_kernel(
    const float* __restrict__ km, _Float16* __restrict__ Fr,
    _Float16* __restrict__ Fi, float* __restrict__ m1,
    float* __restrict__ m2, float* __restrict__ gf) {
  const int idx = blockIdx.x * 256 + threadIdx.x;
  const int j = idx >> 8, k = idx & 255;
  // F[j][k] = exp(-2*pi*i*j*k/256)
  const int p = (j * k) & 255;
  const float ang = -6.28318530717958647692f * (float)p * (1.0f / 256.0f);
  float s, c;
  sincosf(ang, &s, &c);
  Fr[idx] = (_Float16)c;
  Fi[idx] = (_Float16)s;
  // trker transposed (frft2d orientation): softplus(km[k][j]) clipped
  const float v = km[k * N_DIM + j];
  const float sp = (v > 20.f) ? v : log1pf(expf(v));
  const float tr = fminf(fmaxf(sp, 1e-3f), 10.f);
  const float Xc = (float)(k + 1), Yc = (float)(j + 1);
  const float denom = sqrtf(Xc * Xc + Yc * Yc) + 1e-6f;
  m1[idx] = -(Yc / denom) * tr;  // chain producing R1 (y-filter after transposition)
  m2[idx] = -(Xc / denom) * tr;  // chain producing R2
  if (idx < NSLICE) gf[idx] = 0.f;
}

// ---------------------------------------------------------------------------
// Pointwise: A_boost, P, theta + per-slice sum of A_boost (for gf).
// ---------------------------------------------------------------------------
__global__ void __launch_bounds__(256) pointwise_apt(
    const float* __restrict__ x, const float* __restrict__ R1,
    const float* __restrict__ R2, const float* __restrict__ ampw,
    const float* __restrict__ ampb, float* __restrict__ Ab,
    float* __restrict__ P, float* __restrict__ Th, float* __restrict__ gf) {
  const int s = blockIdx.y;
  const size_t base = (size_t)s * SLICE + blockIdx.x * 1024 + threadIdx.x * 4;
  const float aw = ampw[0], ab = ampb[0];
  float lsum = 0.f;
#pragma unroll
  for (int q = 0; q < 4; q++) {
    const size_t e = base + q;
    const float xv = x[e], r1 = R1[e], r2 = R2[e];
    const float a = sqrtf(fmaxf(xv * xv + r1 * r1 + r2 * r2, 0.f) + 1e-6f);
    const float boost = a + fmaxf(a * aw + ab, 0.f);
    Ab[e] = boost;
    P[e] = atan2f(sqrtf(r1 * r1 + r2 * r2) + 1e-6f, xv + 1e-6f);
    Th[e] = atan2f(r2, r1 + 1e-6f);
    lsum += boost;
  }
  __shared__ float red[256];
  red[threadIdx.x] = lsum;
  __syncthreads();
  for (int off = 128; off; off >>= 1) {
    if (threadIdx.x < off) red[threadIdx.x] += red[threadIdx.x + off];
    __syncthreads();
  }
  if (threadIdx.x == 0) atomicAdd(&gf[s], red[0]);
}

// ---------------------------------------------------------------------------
// Fusion: 3x3 (dilated) convs + BN + gating MLP + final 1x1 over 6 channels.
// ---------------------------------------------------------------------------
__device__ __forceinline__ float conv3x3(const float* __restrict__ img, int i,
                                         int j, const float* __restrict__ w,
                                         float bias, int d) {
  float acc = bias;
#pragma unroll
  for (int u = 0; u < 3; u++) {
    const int ii = i + (u - 1) * d;
#pragma unroll
    for (int v = 0; v < 3; v++) {
      const int jj = j + (v - 1) * d;
      const float val = (ii >= 0 && ii < N_DIM && jj >= 0 && jj < N_DIM)
                            ? img[ii * N_DIM + jj] : 0.f;
      acc = fmaf(w[u * 3 + v], val, acc);
    }
  }
  return acc;
}

__global__ void __launch_bounds__(256) fusion_kernel(
    const float* __restrict__ Ab, const float* __restrict__ P,
    const float* __restrict__ Th, const float* __restrict__ gf,
    const float* __restrict__ tw, const float* __restrict__ tb,
    const float* __restrict__ pw, const float* __restrict__ pb,
    const float* __restrict__ pg, const float* __restrict__ pbeta,
    const float* __restrict__ pm, const float* __restrict__ pv,
    const float* __restrict__ w1, const float* __restrict__ b1,
    const float* __restrict__ w2, const float* __restrict__ b2,
    const float* __restrict__ w3, const float* __restrict__ b3,
    const float* __restrict__ fc1w, const float* __restrict__ fc1b,
    const float* __restrict__ fc2w, const float* __restrict__ fc2b,
    const float* __restrict__ fw, const float* __restrict__ fb,
    const float* __restrict__ fg, const float* __restrict__ fbeta,
    const float* __restrict__ fm, const float* __restrict__ fv,
    float* __restrict__ out) {
  const int s = blockIdx.y;
  const float* A = Ab + (size_t)s * SLICE;
  const float* Pp = P + (size_t)s * SLICE;
  const float* Tt = Th + (size_t)s * SLICE;
  const int pix = blockIdx.x * 256 + threadIdx.x;
  const int i = pix >> 8, j = pix & 255;

  // per-slice gating scalar
  const float g = gf[s] * (1.0f / 65536.0f);
  float z = fc2b[0];
#pragma unroll
  for (int q = 0; q < 16; q++) {
    const float h = fmaxf(g * fc1w[q] + fc1b[q], 0.f);
    z = fmaf(h, fc2w[q], z);
  }
  const float gw = 1.f / (1.f + expf(-z));

  const float tc = Tt[pix];
  const float gt = tc * (1.f / (1.f + expf(-conv3x3(Tt, i, j, tw, tb[0], 1))));
  const float pinv = pg[0] * rsqrtf(pv[0] + 1e-5f);
  const float ppr =
      fmaxf((conv3x3(Pp, i, j, pw, pb[0], 1) - pm[0]) * pinv + pbeta[0], 0.f);
  const float m1 = conv3x3(A, i, j, w1, b1[0], 1);
  const float m2 = conv3x3(A, i, j, w2, b2[0], 2);
  const float m3 = conv3x3(A, i, j, w3, b3[0], 3);
  const float ac = A[pix];

  float fpre = fb[0];
  fpre = fmaf(fw[0], ac, fpre);
  fpre = fmaf(fw[1], ppr, fpre);
  fpre = fmaf(fw[2], gt, fpre);
  fpre = fmaf(fw[3], m1 * gw, fpre);
  fpre = fmaf(fw[4], m2 * gw, fpre);
  fpre = fmaf(fw[5], m3 * gw, fpre);
  const float finv = fg[0] * rsqrtf(fv[0] + 1e-5f);
  const float fused = (fpre - fm[0]) * finv + fbeta[0];
  out[(size_t)s * SLICE + pix] = fminf(fmaxf(fused, 0.f), 1.f);
}

// ---------------------------------------------------------------------------
extern "C" void kernel_launch(void* const* d_in, const int* in_sizes, int n_in,
                              void* d_out, int out_size, void* d_ws,
                              size_t ws_size, hipStream_t stream) {
  (void)in_sizes; (void)n_in; (void)out_size; (void)ws_size;
  const float* x        = (const float*)d_in[0];
  const float* km       = (const float*)d_in[1];
  const float* theta_w  = (const float*)d_in[2];
  const float* theta_b  = (const float*)d_in[3];
  const float* phase_w  = (const float*)d_in[4];
  const float* phase_b  = (const float*)d_in[5];
  const float* phase_g  = (const float*)d_in[6];
  const float* phase_be = (const float*)d_in[7];
  const float* phase_m  = (const float*)d_in[8];
  const float* phase_v  = (const float*)d_in[9];
  const float* amp_w    = (const float*)d_in[10];
  const float* amp_b    = (const float*)d_in[11];
  const float* ms_w1    = (const float*)d_in[12];
  const float* ms_b1    = (const float*)d_in[13];
  const float* ms_w2    = (const float*)d_in[14];
  const float* ms_b2    = (const float*)d_in[15];
  const float* ms_w3    = (const float*)d_in[16];
  const float* ms_b3    = (const float*)d_in[17];
  const float* fc1_w    = (const float*)d_in[18];
  const float* fc1_b    = (const float*)d_in[19];
  const float* fc2_w    = (const float*)d_in[20];
  const float* fc2_b    = (const float*)d_in[21];
  const float* fin_w    = (const float*)d_in[22];
  const float* fin_b    = (const float*)d_in[23];
  const float* fin_g    = (const float*)d_in[24];
  const float* fin_be   = (const float*)d_in[25];
  const float* fin_m    = (const float*)d_in[26];
  const float* fin_v    = (const float*)d_in[27];

  // ---- workspace layout (aliased; total ~321 MB) ----
  char* ws = (char*)d_ws;
  const size_t MB = 1ull << 20;
  _Float16* Fr = (_Float16*)(ws);
  _Float16* Fi = (_Float16*)(ws + 131072);
  float* m1    = (float*)(ws + 262144);
  float* m2    = (float*)(ws + 524288);
  float* gf    = (float*)(ws + 786432);
  const size_t RB = 1 * MB;
  _Float16* T1r = (_Float16*)(ws + RB);             // region1 (64 MB)
  _Float16* T1i = (_Float16*)(ws + RB + 32 * MB);
  _Float16* Sr  = (_Float16*)(ws + RB + 64 * MB);   // region2 (64 MB)
  _Float16* Si  = (_Float16*)(ws + RB + 96 * MB);
  _Float16* W1r = T1r;                              // alias region1 (T dead)
  _Float16* W1i = T1i;
  _Float16* W2r = (_Float16*)(ws + RB + 128 * MB);  // region3 (64 MB)
  _Float16* W2i = (_Float16*)(ws + RB + 160 * MB);
  float* R1 = (float*)(ws + RB + 64 * MB);          // alias region2 (S dead)
  float* R2 = (float*)(ws + RB + 192 * MB);         // region4 (64 MB)
  float* Ab = (float*)(ws + RB);                    // alias region1 (W1 dead)
  float* Pm = (float*)(ws + RB + 128 * MB);         // alias region3 (W2 dead)
  float* Th = (float*)(ws + RB + 256 * MB);         // region5 (64 MB)

  // 0) F matrix, spectral maps, zero gf
  init_kernel<<<256, 256, 0, stream>>>(km, Fr, Fi, m1, m2, gf);

  const dim3 g(2, 2, NSLICE);
  // 1) T = F @ x^T          (= (xF)^T)
  dft_gemm<0><<<g, 256, 0, stream>>>(Fr, Fi, x, nullptr, nullptr, nullptr,
                                     T1r, T1i, nullptr);
  // 2) S = F @ T^T = F x F  (natural orientation)
  dft_gemm<1><<<g, 256, 0, stream>>>(Fr, Fi, nullptr, T1r, T1i, nullptr,
                                     Sr, Si, nullptr);
  // 3) W_d = conjF @ (m_d (.) S)^T   (both chains read S before it is reused)
  dft_gemm<2><<<g, 256, 0, stream>>>(Fr, Fi, nullptr, Sr, Si, m1,
                                     W1r, W1i, nullptr);
  dft_gemm<2><<<g, 256, 0, stream>>>(Fr, Fi, nullptr, Sr, Si, m2,
                                     W2r, W2i, nullptr);
  // 4) R_d = |conjF @ W_d^T| / 65536 + 1e-6
  dft_gemm<3><<<g, 256, 0, stream>>>(Fr, Fi, nullptr, W1r, W1i, nullptr,
                                     nullptr, nullptr, R1);
  dft_gemm<3><<<g, 256, 0, stream>>>(Fr, Fi, nullptr, W2r, W2i, nullptr,
                                     nullptr, nullptr, R2);
  // 5) A_boost / P / theta + per-slice means
  pointwise_apt<<<dim3(64, NSLICE), 256, 0, stream>>>(x, R1, R2, amp_w, amp_b,
                                                      Ab, Pm, Th, gf);
  // 6) fusion -> output
  fusion_kernel<<<dim3(256, NSLICE), 256, 0, stream>>>(
      Ab, Pm, Th, gf, theta_w, theta_b, phase_w, phase_b, phase_g, phase_be,
      phase_m, phase_v, ms_w1, ms_b1, ms_w2, ms_b2, ms_w3, ms_b3, fc1_w, fc1_b,
      fc2_w, fc2_b, fin_w, fin_b, fin_g, fin_be, fin_m, fin_v, (float*)d_out);
}
